// LoR_VTranslution_4724464025985
// MI455X (gfx1250) — matrix operand order repaired and checked by reading, not compiled, not run on hardware
//
#include <hip/hip_runtime.h>
#include <stdint.h>

// ---------------------------------------------------------------------------
// MI455X (gfx1250, wave32) implementation.
// All matrix contractions use v_wmma_f32_16x16x32_bf16 (bf16 in, fp32 acc).
// ~25 GFLOP total, ~45MB unique bytes -> compute bound; Wv2 (bf16, 16.8MB)
// stays resident in the 192MB L2 across the rel-loop of the translution term.
// The translution kernel stages u[b] (128KB) into LDS once per block via
// global_load_async_to_lds_b128 (ASYNCcnt) so 8 waves share it with zero
// redundant global traffic.
// ---------------------------------------------------------------------------

typedef __attribute__((ext_vector_type(16))) __bf16 v16bf;
typedef __attribute__((ext_vector_type(8)))  __bf16 v8bf;
typedef __attribute__((ext_vector_type(8)))  float  v8f;

union BF16x16 { v16bf v; v8bf h[2]; __bf16 e[16]; };
union F32x8   { v8f   v; float e[8]; };

__device__ __forceinline__ v8f zero_v8f() {
  v8f z = {0.f,0.f,0.f,0.f,0.f,0.f,0.f,0.f};
  return z;
}

// A-fragment (16x32 bf16, MxK), ISA 7.12.2: lane l -> row (l&15);
// elements 0..7 = K 8*(l>>4)+e ; elements 8..15 = K 16+8*(l>>4)+(e-8)
__device__ __forceinline__ v16bf load_a_frag(const __bf16* __restrict__ A, int lda) {
  int l = threadIdx.x & 31;
  const __bf16* p = A + (long)(l & 15) * lda + ((l >> 4) << 3);
  BF16x16 r;
  r.h[0] = *(const v8bf*)p;
  r.h[1] = *(const v8bf*)(p + 16);
  return r.v;
}

// B-fragment (32x16 bf16, KxN): lane l -> col N=(l&15); elements 0..7 = K 8*(l>>4)+e ; 8..15 = K 16+8*(l>>4)+(e-8)
__device__ __forceinline__ v16bf load_b_frag(const __bf16* __restrict__ B, int ldb) {
  int l = threadIdx.x & 31;
  const __bf16* p = B + (long)((l >> 4) << 3) * ldb + (l & 15);
  BF16x16 r;
#pragma unroll
  for (int i = 0; i < 8; ++i) {
    r.e[i]     = p[(long)i * ldb];
    r.e[8 + i] = p[(long)(16 + i) * ldb];
  }
  return r.v;
}

// One 16-row x (NT*16)-col output strip, contraction over K (multiple of 32).
// NT independent accumulators let the scheduler interleave WMMAs (hazard cover).
template <int NT>
__device__ __forceinline__ void mma_strip(const __bf16* __restrict__ Ab, int lda,
                                          const __bf16* __restrict__ Bb, int ldb,
                                          int K, F32x8* acc) {
  for (int k = 0; k < K; k += 32) {
    v16bf af = load_a_frag(Ab + k, lda);
#pragma unroll
    for (int j = 0; j < NT; ++j) {
      v16bf bfr = load_b_frag(Bb + (long)k * ldb + j * 16, ldb);
      acc[j].v = __builtin_amdgcn_wmma_f32_16x16x32_bf16(
          false, af, false, bfr, (short)0, acc[j].v, false, false);
    }
  }
}

// ---------------------------------------------------------------------------
__global__ void k_cast_f32_bf16(const float* __restrict__ in, __bf16* __restrict__ out, int n) {
  int i = blockIdx.x * blockDim.x + threadIdx.x;
  if (i < n) out[i] = (__bf16)in[i];
}

// LayerNorm over 512 columns, one block per row, writes bf16.
__global__ void k_layernorm(const float* __restrict__ x, const float* __restrict__ g,
                            const float* __restrict__ b, __bf16* __restrict__ xn) {
  int row = blockIdx.x;
  const float* xr = x + (long)row * 512;
  float s = 0.f, s2 = 0.f;
  for (int i = threadIdx.x; i < 512; i += 256) { float v = xr[i]; s += v; s2 += v * v; }
#pragma unroll
  for (int off = 16; off; off >>= 1) { s += __shfl_xor(s, off, 32); s2 += __shfl_xor(s2, off, 32); }
  __shared__ float sa[8], sb[8];
  int w = threadIdx.x >> 5;
  if ((threadIdx.x & 31) == 0) { sa[w] = s; sb[w] = s2; }
  __syncthreads();
  float ts = 0.f, ts2 = 0.f;
#pragma unroll
  for (int i = 0; i < 8; ++i) { ts += sa[i]; ts2 += sb[i]; }
  float mu  = ts * (1.f / 512.f);
  float var = ts2 * (1.f / 512.f) - mu * mu;
  float r   = rsqrtf(var + 1e-5f);
  for (int i = threadIdx.x; i < 512; i += 256)
    xn[(long)row * 512 + i] = (__bf16)(((xr[i] - mu) * r) * g[i] + b[i]);
}

// Generic wave-per-(16x128)-strip bf16 GEMM. EPI: 0=store bf16,
// 1=store bf16((acc+extra)*0.5), 2=store f32 acc+bias[col].
template <int EPI>
__global__ void k_gemm(const __bf16* __restrict__ A, const __bf16* __restrict__ B,
                       void* __restrict__ C, const float* __restrict__ extra,
                       int N, int K, int lda, int ldb, int ldc) {
  int wid = blockIdx.x * (blockDim.x >> 5) + (threadIdx.x >> 5);
  int nw = N >> 7; // strips of 128 columns
  int mt = wid / nw, nt = wid % nw;
  const __bf16* Ab = A + (long)mt * 16 * lda;
  const __bf16* Bb = B + (long)nt * 128;
  F32x8 acc[8];
#pragma unroll
  for (int j = 0; j < 8; ++j) acc[j].v = zero_v8f();
  mma_strip<8>(Ab, lda, Bb, ldb, K, acc);
  int l = threadIdx.x & 31, cl = l & 15, hl = l >> 4;
#pragma unroll
  for (int j = 0; j < 8; ++j) {
#pragma unroll
    for (int v = 0; v < 8; ++v) {
      long row = (long)mt * 16 + hl * 8 + v;
      long col = (long)nt * 128 + j * 16 + cl;
      long idx = row * ldc + col;
      if (EPI == 0)      ((__bf16*)C)[idx] = (__bf16)acc[j].e[v];
      else if (EPI == 1) ((__bf16*)C)[idx] = (__bf16)((acc[j].e[v] + extra[idx]) * 0.5f);
      else               ((float*)C)[idx]  = acc[j].e[v] + extra[col];
    }
  }
}

// kT[b][h][dh][m] = k (cols 512..1023 of qkv), so QK^T B-fragments are contiguous.
__global__ void k_transpose_k(const __bf16* __restrict__ qkv, __bf16* __restrict__ kT) {
  int idx = blockIdx.x * blockDim.x + threadIdx.x; // 4*8*64*512 = 1M
  int m  = idx & 511;
  int dh = (idx >> 9) & 63;
  int h  = (idx >> 15) & 7;
  int b  = idx >> 18;
  kT[idx] = qkv[((long)(b * 512 + m)) * 1536 + 512 + h * 64 + dh];
}

// Block = (b,h, 16-row n-tile); 8 waves each own a 64-wide m strip.
// WMMA QK^T -> fp32 causal softmax (shfl_xor half-wave + LDS cross-wave) -> bf16 attn.
__global__ void k_attention(const __bf16* __restrict__ qkv, const __bf16* __restrict__ kT,
                            __bf16* __restrict__ attn) {
  int blk = blockIdx.x;
  int ntile = blk & 31;
  int bh = blk >> 5;
  int h = bh & 7;
  int n0 = ntile << 4;
  int w = threadIdx.x >> 5;
  int l = threadIdx.x & 31, cl = l & 15, hl = l >> 4;

  const __bf16* qbase = qkv + ((long)(bh >> 3) * 512 + n0) * 1536 + h * 64;
  const __bf16* kbase = kT + (long)bh * 64 * 512 + w * 64;
  F32x8 acc[4];
#pragma unroll
  for (int j = 0; j < 4; ++j) acc[j].v = zero_v8f();
  mma_strip<4>(qbase, 1536, kbase, 512, 64, acc);

  const float scale = 0.125f; // 1/sqrt(64)
#pragma unroll
  for (int t = 0; t < 4; ++t)
#pragma unroll
    for (int v = 0; v < 8; ++v) {
      int m = (w << 6) + (t << 4) + cl;
      int n = n0 + v + (hl << 3);
      float d = acc[t].e[v] * scale;
      acc[t].e[v] = (m <= n) ? d : -3.0e38f;
    }

  __shared__ float red[8][16];
  float rmax[8], rsum[8];
#pragma unroll
  for (int v = 0; v < 8; ++v) {
    float x = fmaxf(fmaxf(acc[0].e[v], acc[1].e[v]), fmaxf(acc[2].e[v], acc[3].e[v]));
#pragma unroll
    for (int off = 8; off; off >>= 1) x = fmaxf(x, __shfl_xor(x, off, 32));
    rmax[v] = x;
  }
  if (cl == 0) {
#pragma unroll
    for (int v = 0; v < 8; ++v) red[w][(hl << 3) + v] = rmax[v];
  }
  __syncthreads();
#pragma unroll
  for (int v = 0; v < 8; ++v) {
    float x = -3.4e38f;
#pragma unroll
    for (int i = 0; i < 8; ++i) x = fmaxf(x, red[i][(hl << 3) + v]);
    rmax[v] = x;
  }
  __syncthreads();
#pragma unroll
  for (int v = 0; v < 8; ++v) {
    float s = 0.f;
#pragma unroll
    for (int t = 0; t < 4; ++t) {
      float e = __expf(acc[t].e[v] - rmax[v]);
      acc[t].e[v] = e;
      s += e;
    }
#pragma unroll
    for (int off = 8; off; off >>= 1) s += __shfl_xor(s, off, 32);
    rsum[v] = s;
  }
  if (cl == 0) {
#pragma unroll
    for (int v = 0; v < 8; ++v) red[w][(hl << 3) + v] = rsum[v];
  }
  __syncthreads();
#pragma unroll
  for (int v = 0; v < 8; ++v) {
    float s = 0.f;
#pragma unroll
    for (int i = 0; i < 8; ++i) s += red[i][(hl << 3) + v];
    rsum[v] = 1.f / s;
  }
  __bf16* abase = attn + (long)bh * 512 * 512;
#pragma unroll
  for (int t = 0; t < 4; ++t)
#pragma unroll
    for (int v = 0; v < 8; ++v) {
      int m = (w << 6) + (t << 4) + cl;
      int n = n0 + v + (hl << 3);
      abase[(long)n * 512 + m] = (__bf16)(acc[t].e[v] * rsum[v]);
    }
}

// out1 = attn @ v1 per (b,h): M=512, K=512, N=64 -> fp32 into comb.
__global__ void k_out1(const __bf16* __restrict__ attn, const __bf16* __restrict__ qkv,
                       float* __restrict__ comb) {
  int bh = blockIdx.y;
  int h = bh & 7;
  const __bf16* A = attn + (long)bh * 512 * 512;
  const __bf16* B = qkv + 1024 + h * 64;
  float* C = comb + (long)(bh >> 3) * 512 * 512 + h * 64;
  int mt = blockIdx.x * 8 + (threadIdx.x >> 5); // 0..31
  F32x8 acc[4];
#pragma unroll
  for (int j = 0; j < 4; ++j) acc[j].v = zero_v8f();
  mma_strip<4>(A + (long)mt * 16 * 512, 512, B, 1536, 512, acc);
  int l = threadIdx.x & 31, cl = l & 15, hl = l >> 4;
#pragma unroll
  for (int j = 0; j < 4; ++j)
#pragma unroll
    for (int v = 0; v < 8; ++v)
      C[(long)(mt * 16 + hl * 8 + v) * 512 + j * 16 + cl] = acc[j].e[v];
}

// Translution term, restructured by rel = n-m into WMMA chains:
// out2[b,h,n,dr] = sum_rel ( attn[b,h,n,n-rel] * u[b,n-rel,:] ) @ Wv2[rel,:,h*16+dr]
// Block = (b, 16-row n-tile); wave w = head h; acc is one 16x16 tile.
// u[b] (512x128 bf16 = 128KB) is staged to LDS once per block with
// global_load_async_to_lds_b128 (ASYNCcnt path), shared by all 8 waves.
__global__ void k_translution(const __bf16* __restrict__ attn, const __bf16* __restrict__ u,
                              const __bf16* __restrict__ wv2, __bf16* __restrict__ out2) {
  extern __shared__ __align__(16) __bf16 s_u[]; // 512*128 bf16 = 128 KB dynamic LDS
  int b = blockIdx.x >> 5;
  int ntile = blockIdx.x & 31;
  int n0 = ntile << 4;
  int h = threadIdx.x >> 5;
  int l = threadIdx.x & 31, arow = l & 15, kh = l >> 4;
  int n = n0 + arow;

  // Async-stage u[b] into LDS: 8192 x 16B chunks over 256 threads.
  {
    unsigned lds_base = (unsigned)(uintptr_t)&s_u[0];
    unsigned long long gbase = (unsigned long long)(uintptr_t)(u + (long)b * 512 * 128);
    for (int i = threadIdx.x; i < 8192; i += 256) {
      unsigned loff = lds_base + (unsigned)i * 16u;
      unsigned long long ga = gbase + (unsigned long long)i * 16ull;
      asm volatile("global_load_async_to_lds_b128 %0, %1, off"
                   :: "v"(loff), "v"(ga) : "memory");
    }
    asm volatile("s_wait_asynccnt 0" ::: "memory");
  }
  __syncthreads();

  const __bf16* attn_bh = attn + (long)(b * 8 + h) * 512 * 512;

  F32x8 acc;
  acc.v = zero_v8f();
  int relmax = n0 + 15;
  for (int rel = 0; rel <= relmax; ++rel) {
    int m = n - rel;
    float a = 0.f;
    int mc = m < 0 ? 0 : m;
    if (m >= 0) a = (float)attn_bh[(long)n * 512 + m];
    const __bf16* urow = s_u + (long)mc * 128; // LDS reads (ds_load_b128)
    const __bf16* wrow = wv2 + (long)rel * 16384 + h * 16;
    __builtin_prefetch(wrow + 16384 + (long)l * 128, 0, 3); // next rel's Wv2 slice
#pragma unroll
    for (int kc = 0; kc < 4; ++kc) {
      // A fragment: row (l&15) = diag-scaled u[n-rel, kc*32 .. kc*32+31]
      v8bf lo = *(const v8bf*)(urow + kc * 32 + (kh << 3));
      v8bf hi = *(const v8bf*)(urow + kc * 32 + 16 + (kh << 3));
      BF16x16 af;
#pragma unroll
      for (int i = 0; i < 8; ++i) {
        af.e[i]     = (__bf16)(a * (float)lo[i]);
        af.e[8 + i] = (__bf16)(a * (float)hi[i]);
      }
      // B fragment: lane l -> col (l&15) of head h, rows d = kc*32 + A-fragment K order
      const __bf16* bp = wrow + (long)(kc * 32 + (kh << 3)) * 128 + (l & 15);
      BF16x16 bf_;
#pragma unroll
      for (int i = 0; i < 8; ++i) {
        bf_.e[i]     = bp[(long)i * 128];
        bf_.e[8 + i] = bp[(long)(16 + i) * 128];
      }
      acc.v = __builtin_amdgcn_wmma_f32_16x16x32_bf16(
          false, af.v, false, bf_.v, (short)0, acc.v, false, false);
    }
  }
  int cl = l & 15;
#pragma unroll
  for (int v = 0; v < 8; ++v)
    out2[(long)(b * 512 + n0 + (kh << 3) + v) * 128 + h * 16 + cl] = (__bf16)acc.e[v];
}

// ---------------------------------------------------------------------------
extern "C" void kernel_launch(void* const* d_in, const int* in_sizes, int n_in,
                              void* d_out, int out_size, void* d_ws, size_t ws_size,
                              hipStream_t stream) {
  const float* x    = (const float*)d_in[0];
  const float* ln_g = (const float*)d_in[1];
  const float* ln_b = (const float*)d_in[2];
  const float* Wqkv = (const float*)d_in[3];
  const float* Wv1  = (const float*)d_in[4];
  const float* Wv2  = (const float*)d_in[5];
  const float* Wv3  = (const float*)d_in[6];
  const float* Wout = (const float*)d_in[7];
  const float* bout = (const float*)d_in[8];
  float* out = (float*)d_out;

  char* p = (char*)d_ws;
  auto bump = [&](size_t bytes) -> void* {
    void* r = (void*)p;
    p += (bytes + 255) & ~(size_t)255;
    return r;
  };
  __bf16* xn_bf   = (__bf16*)bump((size_t)2048 * 512 * 2);
  __bf16* Wqkv_bf = (__bf16*)bump((size_t)512 * 1536 * 2);
  __bf16* Wv1_bf  = (__bf16*)bump((size_t)512 * 128 * 2);
  __bf16* Wv2_bf  = (__bf16*)bump((size_t)512 * 128 * 128 * 2);
  __bf16* Wv3_bf  = (__bf16*)bump((size_t)128 * 512 * 2);
  __bf16* Wout_bf = (__bf16*)bump((size_t)512 * 512 * 2);
  __bf16* qkv_bf  = (__bf16*)bump((size_t)2048 * 1536 * 2);
  __bf16* u_bf    = (__bf16*)bump((size_t)2048 * 128 * 2);
  __bf16* kT_bf   = (__bf16*)bump((size_t)32 * 64 * 512 * 2);
  __bf16* attn_bf = (__bf16*)bump((size_t)32 * 512 * 512 * 2);
  float*  comb    = (float*)bump((size_t)2048 * 512 * 4);
  __bf16* out2_bf = (__bf16*)bump((size_t)2048 * 128 * 2);
  __bf16* comb_bf = (__bf16*)bump((size_t)2048 * 512 * 2);

  auto cast = [&](const float* src, __bf16* dst, int n) {
    k_cast_f32_bf16<<<(n + 255) / 256, 256, 0, stream>>>(src, dst, n);
  };
  cast(Wqkv, Wqkv_bf, 512 * 1536);
  cast(Wv1,  Wv1_bf,  512 * 128);
  cast(Wv2,  Wv2_bf,  512 * 128 * 128);
  cast(Wv3,  Wv3_bf,  128 * 512);
  cast(Wout, Wout_bf, 512 * 512);

  k_layernorm<<<2048, 256, 0, stream>>>(x, ln_g, ln_b, xn_bf);

  // qkv = xn @ Wqkv : M=2048 N=1536 K=512 -> 1536 waves (16x128 strips)
  k_gemm<0><<<192, 256, 0, stream>>>(xn_bf, Wqkv_bf, (void*)qkv_bf, nullptr,
                                     1536, 512, 512, 1536, 1536);
  // u = xn @ Wv1 : M=2048 N=128 K=512 -> 128 waves
  k_gemm<0><<<16, 256, 0, stream>>>(xn_bf, Wv1_bf, (void*)u_bf, nullptr,
                                    128, 512, 512, 128, 128);

  k_transpose_k<<<4096, 256, 0, stream>>>(qkv_bf, kT_bf);
  k_attention<<<1024, 256, 0, stream>>>(qkv_bf, kT_bf, attn_bf);
  k_out1<<<dim3(4, 32), 256, 0, stream>>>(attn_bf, qkv_bf, comb);
  k_translution<<<128, 256, 131072, stream>>>(attn_bf, u_bf, Wv2_bf, out2_bf);

  // comb_bf = (out1 + out2 @ Wv3) * 0.5 : M=2048 N=512 K=128 -> 512 waves
  k_gemm<1><<<64, 256, 0, stream>>>(out2_bf, Wv3_bf, (void*)comb_bf, comb,
                                    512, 128, 128, 512, 512);
  // out = comb_bf @ Wout + bout : M=2048 N=512 K=512 -> 512 waves
  k_gemm<2><<<64, 256, 0, stream>>>(comb_bf, Wout_bf, (void*)out, bout,
                                    512, 512, 512, 512, 512);
}
